// CrossMultiheadAttention_23029614641617
// MI455X (gfx1250) — matrix operand order repaired and checked by reading, not compiled, not run on hardware
//
#include <hip/hip_runtime.h>
#include <hip/hip_bf16.h>

// Problem constants (B,T,S,E,H,D) = (2,1024,1024,1024,16,64)
#define BB 2
#define TT 1024
#define SS 1024
#define EE 1024
#define HH 16
#define DD 64
#define SCALING 0.125f   // D^-0.5 = 1/8

typedef __attribute__((ext_vector_type(16))) __bf16 v16bf;
typedef __attribute__((ext_vector_type(8)))  float  v8f;

static __device__ __forceinline__ v8f wmma_bf16(v16bf a, v16bf b, v8f c) {
  // D = A(16x32 bf16) * B(32x16 bf16) + C(16x16 f32)
  return __builtin_amdgcn_wmma_f32_16x16x32_bf16(false, a, false, b, (short)0, c,
                                                 false, false);
}

// ---- Fragment builders (ISA 7.12.2 row/column lanes, contiguous k order, wave32) -------------------
// A (16x32, 16-bit): lane l holds row (l&15); K=(l>>4)*16+e (contiguous,
// matching the B fragment order).
static __device__ __forceinline__ v16bf load_a_bf16(const __bf16* p, int ld, int lane) {
  const __bf16* r = p + (size_t)(lane & 15) * ld + ((lane >> 4) << 4);
  v16bf a;
#pragma unroll
  for (int e = 0; e < 8; ++e) { a[e] = r[e]; a[e + 8] = r[8 + e]; }
  return a;
}
static __device__ __forceinline__ v16bf load_a_f32(const float* p, int ld, int lane) {
  const float* r = p + (size_t)(lane & 15) * ld + ((lane >> 4) << 4);
  v16bf a;
#pragma unroll
  for (int e = 0; e < 8; ++e) { a[e] = (__bf16)r[e]; a[e + 8] = (__bf16)r[8 + e]; }
  return a;
}
// B (32x16, 16-bit) built as transpose of a row-major [N x K] matrix:
// memory[n*ld + k] supplies B(k,n). Lane l: n=(l&15), K=(l>>4)*16+e -> 16
// contiguous elements per lane (vectorizable global_load).
static __device__ __forceinline__ v16bf load_bt_bf16(const __bf16* p, int ld, int lane) {
  const __bf16* r = p + (size_t)(lane & 15) * ld + ((lane >> 4) << 4);
  v16bf b;
#pragma unroll
  for (int e = 0; e < 16; ++e) b[e] = r[e];
  return b;
}
static __device__ __forceinline__ v16bf load_bt_f32(const float* p, int ld, int lane) {
  const float* r = p + (size_t)(lane & 15) * ld + ((lane >> 4) << 4);
  v16bf b;
#pragma unroll
  for (int e = 0; e < 16; ++e) b[e] = (__bf16)r[e];
  return b;
}

// ---- Kernel 1: input projections  out = (X @ W^T + bias) * scale ----------
// X: [B*1024, E] fp32 ; W: [E,E] fp32 (torch Linear, row = out feature).
// Output bf16, head-split [b,h,t,d] (transpose_out=0) or [b,h,d,t] (=1, for V).
// Grid: 512 blocks x 256 thr; 8 waves/WG -> 64x64 tile; wave = 16x32 subtile.
__global__ __launch_bounds__(256)
void proj_kernel(const float* __restrict__ X, const float* __restrict__ W,
                 const float* __restrict__ bias, __bf16* __restrict__ out,
                 float scale, int transpose_out) {
  const int lane = threadIdx.x & 31;
  const int wv   = threadIdx.x >> 5;
  const int mrow = (blockIdx.x >> 4) * 64 + (wv & 3) * 16;
  const int ncol = (blockIdx.x & 15) * 64 + (wv >> 2) * 32;

  v8f acc0 = {}, acc1 = {};
  const float* Xrow = X + (size_t)mrow * EE;
  const float* W0   = W + (size_t)ncol * EE;
  const float* W1   = W + (size_t)(ncol + 16) * EE;
  for (int kk = 0; kk < EE; kk += 32) {
    v16bf a  = load_a_f32 (Xrow + kk, EE, lane);
    v16bf b0 = load_bt_f32(W0  + kk, EE, lane);
    v16bf b1 = load_bt_f32(W1  + kk, EE, lane);
    acc0 = wmma_bf16(a, b0, acc0);
    acc1 = wmma_bf16(a, b1, acc1);
  }
  const int nlo = lane & 15, half = lane >> 4;
#pragma unroll
  for (int r = 0; r < 8; ++r) {
    int g = mrow + r + 8 * half;          // global row: b*1024 + t
    int b = g >> 10, t = g & 1023;
    int o0 = ncol + nlo, o1 = o0 + 16;
    float v0 = (acc0[r] + bias[o0]) * scale;
    float v1 = (acc1[r] + bias[o1]) * scale;
    int h0 = o0 >> 6, d0 = o0 & 63;
    int h1 = o1 >> 6, d1 = o1 & 63;
    if (transpose_out) {
      out[(((size_t)b * HH + h0) * DD + d0) * SS + t] = (__bf16)v0;
      out[(((size_t)b * HH + h1) * DD + d1) * SS + t] = (__bf16)v1;
    } else {
      out[(((size_t)b * HH + h0) * TT + t) * DD + d0] = (__bf16)v0;
      out[(((size_t)b * HH + h1) * TT + t) * DD + d1] = (__bf16)v1;
    }
  }
}

// ---- Kernel 2: fused flash attention --------------------------------------
// One wave owns 16 query rows of one (b,h); loops S in 32-wide chunks:
// scores via 4 WMMAs (K-dim=64), mask->-inf, +bias, online softmax, P->LDS
// (C-layout -> A-layout), PV via 4 WMMAs. Bias (128 MB) is streamed once.
__global__ __launch_bounds__(256)
void attn_kernel(const __bf16* __restrict__ Qh, const __bf16* __restrict__ Kh,
                 const __bf16* __restrict__ Vt, const unsigned char* __restrict__ kpm,
                 const float* __restrict__ bias, __bf16* __restrict__ O) {
  __shared__ __bf16 pbuf[8][16 * 40];          // per-wave 16x32 bf16, stride 40

  const int lane = threadIdx.x & 31;
  const int wv   = threadIdx.x >> 5;
  const int bh   = blockIdx.x >> 3;            // 32 (b,h) pairs
  const int b    = bh >> 4, h = bh & 15;
  const int trow = (blockIdx.x & 7) * 128 + wv * 16;
  const int nlo = lane & 15, half = lane >> 4;
  const float NEG_INF = -__builtin_inff();

  const __bf16* Qbase = Qh + ((size_t)bh * TT + trow) * DD;  // Q pre-scaled
  v16bf aq0 = load_a_bf16(Qbase,      DD, lane);             // K(d) 0..31
  v16bf aq1 = load_a_bf16(Qbase + 32, DD, lane);             // K(d) 32..63

  const __bf16* Kbase = Kh + (size_t)bh * SS * DD;           // [s][d]
  const __bf16* Vbase = Vt + (size_t)bh * DD * SS;           // [d][s]
  const unsigned char* mbase = kpm + (size_t)b * SS;
  const float* biasbase = bias + (size_t)bh * TT * SS;
  __bf16* pw = &pbuf[wv][0];

  v8f acc0 = {}, acc1 = {}, acc2 = {}, acc3 = {};            // O: d 0-15..48-63
  float mrun[8], lrun[8];
#pragma unroll
  for (int r = 0; r < 8; ++r) { mrun[r] = NEG_INF; lrun[r] = 0.f; }

  for (int s0 = 0; s0 < SS; s0 += 32) {
    v8f c0 = {}, c1 = {};
    {
      v16bf bk = load_bt_bf16(Kbase + (size_t)s0 * DD,            DD, lane);
      c0 = wmma_bf16(aq0, bk, c0);
      bk = load_bt_bf16(Kbase + (size_t)s0 * DD + 32,             DD, lane);
      c0 = wmma_bf16(aq1, bk, c0);
      bk = load_bt_bf16(Kbase + (size_t)(s0 + 16) * DD,           DD, lane);
      c1 = wmma_bf16(aq0, bk, c1);
      bk = load_bt_bf16(Kbase + (size_t)(s0 + 16) * DD + 32,      DD, lane);
      c1 = wmma_bf16(aq1, bk, c1);
    }
    const int col0 = s0 + nlo, col1 = col0 + 16;
    const bool mk0 = mbase[col0] != 0;
    const bool mk1 = mbase[col1] != 0;
#pragma unroll
    for (int r = 0; r < 8; ++r) {
      const int t = trow + r + 8 * half;
      const float* bp = biasbase + (size_t)t * SS;
      float v0 = mk0 ? NEG_INF : (c0[r] + bp[col0]);
      float v1 = mk1 ? NEG_INF : (c1[r] + bp[col1]);
      float mv = fmaxf(v0, v1);                 // row max over 32 cols:
      mv = fmaxf(mv, __shfl_xor(mv, 1));        // butterfly within 16-lane half
      mv = fmaxf(mv, __shfl_xor(mv, 2));
      mv = fmaxf(mv, __shfl_xor(mv, 4));
      mv = fmaxf(mv, __shfl_xor(mv, 8));
      float mnew = fmaxf(mrun[r], mv);
      float corr = (mrun[r] == mnew) ? 1.f : __expf(mrun[r] - mnew);
      float p0 = (mnew == NEG_INF) ? 0.f : __expf(v0 - mnew);
      float p1 = (mnew == NEG_INF) ? 0.f : __expf(v1 - mnew);
      float ps = p0 + p1;
      ps += __shfl_xor(ps, 1);
      ps += __shfl_xor(ps, 2);
      ps += __shfl_xor(ps, 4);
      ps += __shfl_xor(ps, 8);
      lrun[r] = lrun[r] * corr + ps;
      mrun[r] = mnew;
      acc0[r] *= corr; acc1[r] *= corr; acc2[r] *= corr; acc3[r] *= corr;
      pw[(r + 8 * half) * 40 + nlo]      = (__bf16)p0;   // C-layout -> row major
      pw[(r + 8 * half) * 40 + 16 + nlo] = (__bf16)p1;
    }
    asm volatile("s_wait_dscnt 0" ::: "memory");           // P visible to wave
    v16bf ap = load_a_bf16(pw, 40, lane);                  // 16x32 A fragment
    acc0 = wmma_bf16(ap, load_bt_bf16(Vbase + (size_t) 0 * SS + s0, SS, lane), acc0);
    acc1 = wmma_bf16(ap, load_bt_bf16(Vbase + (size_t)16 * SS + s0, SS, lane), acc1);
    acc2 = wmma_bf16(ap, load_bt_bf16(Vbase + (size_t)32 * SS + s0, SS, lane), acc2);
    acc3 = wmma_bf16(ap, load_bt_bf16(Vbase + (size_t)48 * SS + s0, SS, lane), acc3);
  }
#pragma unroll
  for (int r = 0; r < 8; ++r) {
    int t = trow + r + 8 * half;
    float inv = lrun[r] > 0.f ? 1.f / lrun[r] : 0.f;
    __bf16* op = O + ((size_t)b * TT + t) * EE + h * DD;   // merge heads
    op[ 0 + nlo] = (__bf16)(acc0[r] * inv);
    op[16 + nlo] = (__bf16)(acc1[r] * inv);
    op[32 + nlo] = (__bf16)(acc2[r] * inv);
    op[48 + nlo] = (__bf16)(acc3[r] * inv);
  }
}

// ---- Kernel 3: output projection  out = A @ Wo^T + bo (fp32 out) ----------
__global__ __launch_bounds__(256)
void oproj_kernel(const __bf16* __restrict__ A, const float* __restrict__ W,
                  const float* __restrict__ bias, float* __restrict__ out) {
  const int lane = threadIdx.x & 31;
  const int wv   = threadIdx.x >> 5;
  const int mrow = (blockIdx.x >> 4) * 64 + (wv & 3) * 16;
  const int ncol = (blockIdx.x & 15) * 64 + (wv >> 2) * 32;
  v8f acc0 = {}, acc1 = {};
  const __bf16* Arow = A + (size_t)mrow * EE;
  const float* W0 = W + (size_t)ncol * EE;
  const float* W1 = W + (size_t)(ncol + 16) * EE;
  for (int kk = 0; kk < EE; kk += 32) {
    v16bf a  = load_a_bf16(Arow + kk, EE, lane);
    v16bf b0 = load_bt_f32(W0 + kk, EE, lane);
    v16bf b1 = load_bt_f32(W1 + kk, EE, lane);
    acc0 = wmma_bf16(a, b0, acc0);
    acc1 = wmma_bf16(a, b1, acc1);
  }
  const int nlo = lane & 15, half = lane >> 4;
#pragma unroll
  for (int r = 0; r < 8; ++r) {
    int g = mrow + r + 8 * half;
    out[(size_t)g * EE + ncol + nlo]      = acc0[r] + bias[ncol + nlo];
    out[(size_t)g * EE + ncol + 16 + nlo] = acc1[r] + bias[ncol + 16 + nlo];
  }
}

extern "C" void kernel_launch(void* const* d_in, const int* in_sizes, int n_in,
                              void* d_out, int out_size, void* d_ws, size_t ws_size,
                              hipStream_t stream) {
  (void)in_sizes; (void)n_in; (void)out_size; (void)ws_size;
  const float* query = (const float*)d_in[0];
  const float* key_  = (const float*)d_in[1];
  const float* value = (const float*)d_in[2];
  const unsigned char* kpm = (const unsigned char*)d_in[3];  // bool mask
  const float* attn_bias = (const float*)d_in[4];
  const float* Wq = (const float*)d_in[5];
  const float* bq = (const float*)d_in[6];
  const float* Wk = (const float*)d_in[7];
  const float* bk = (const float*)d_in[8];
  const float* Wv = (const float*)d_in[9];
  const float* bv = (const float*)d_in[10];
  const float* Wo = (const float*)d_in[11];
  const float* bo = (const float*)d_in[12];

  // Workspace: 4 bf16 buffers of B*H*1024*64 = 2Mi elements (4 MB) each.
  const size_t bytesPer = (size_t)BB * HH * TT * DD * sizeof(__bf16);
  char* ws = (char*)d_ws;
  __bf16* Qh = (__bf16*)(ws + 0 * bytesPer);   // [b,h,t,d], pre-scaled by 1/8
  __bf16* Kh = (__bf16*)(ws + 1 * bytesPer);   // [b,h,s,d]
  __bf16* Vt = (__bf16*)(ws + 2 * bytesPer);   // [b,h,d,s] (transposed)
  __bf16* Ob = (__bf16*)(ws + 3 * bytesPer);   // [b,t,e]

  dim3 blk(256);
  proj_kernel<<<dim3(512), blk, 0, stream>>>(query, Wq, bq, Qh, SCALING, 0);
  proj_kernel<<<dim3(512), blk, 0, stream>>>(key_,  Wk, bk, Kh, 1.0f,    0);
  proj_kernel<<<dim3(512), blk, 0, stream>>>(value, Wv, bv, Vt, 1.0f,    1);
  attn_kernel<<<dim3(256), blk, 0, stream>>>(Qh, Kh, Vt, kpm, attn_bias, Ob);
  oproj_kernel<<<dim3(512), blk, 0, stream>>>(Ob, Wo, bo, (float*)d_out);
}
